// IlluminationGuidedAttention_23622320128073
// MI455X (gfx1250) — matrix operand order repaired and checked by reading, not compiled, not run on hardware
//
#include <hip/hip_runtime.h>

#ifndef __has_builtin
#define __has_builtin(x) 0
#endif
#if __has_builtin(__builtin_amdgcn_global_load_async_to_lds_b128)
#define HAVE_ASYNC_LDS 1
#else
#define HAVE_ASYNC_LDS 0
#endif

// ---------------- types ----------------
typedef __bf16 bf16;
typedef __attribute__((ext_vector_type(16))) __bf16 v16bf;
typedef __attribute__((ext_vector_type(8)))  __bf16 v8bf;
typedef __attribute__((ext_vector_type(8)))  float  v8f;
typedef __attribute__((ext_vector_type(4)))  int    v4i;

#define C_     256
#define H_     256
#define W_     256
#define HW_    65536
#define B_     2
#define HEADS_ 8
#define D_     32
#define WS_    8

// ---------------- async global->LDS helpers (ASYNCcnt path) ----------------
#if HAVE_ASYNC_LDS
typedef __attribute__((address_space(1))) v4i as1_v4i;   // "__device__" int4
typedef __attribute__((address_space(3))) v4i as3_v4i;   // "__shared__" int4

__device__ __forceinline__ void async_copy_b128(const bf16* g, bf16* l) {
  __builtin_amdgcn_global_load_async_to_lds_b128(
      (as1_v4i*)(void*)const_cast<bf16*>(g),
      (as3_v4i*)(void*)l, 0, 0);
}
#endif

__device__ __forceinline__ void async_join() {
#if HAVE_ASYNC_LDS
#if __has_builtin(__builtin_amdgcn_s_wait_asynccnt)
  __builtin_amdgcn_s_wait_asynccnt(0);
#else
  asm volatile("s_wait_asynccnt 0" ::: "memory");
#endif
#endif
}

// ---------------- WMMA fragment helpers (CDNA5 wave32 layouts) ----------------
// A (16x32 bf16): lane m = lane&15; lanes0-15: e0..7->K0..7, e8..15->K16..23
//                 lanes16-31: e0..7->K8..15, e8..15->K24..31
__device__ __forceinline__ v16bf load_a_frag(const bf16* base, int ldm, int lane) {
  int m  = lane & 15;
  int kb = (lane >> 4) << 3;
  const bf16* row = base + m * ldm + kb;
  union { v16bf v; v8bf h[2]; } u;
  u.h[0] = *(const v8bf*)(row);          // ds_load_b128
  u.h[1] = *(const v8bf*)(row + 16);     // ds_load_b128
  return u.v;
}

// B (32x16 bf16) staged transposed as BT[N][K]: lane n = lane&15;
// lanes0-15: e0..7->K0..7, e8..15->K16..23; lanes16-31: e0..7->K8..15, e8..15->K24..31.
__device__ __forceinline__ v16bf load_b_frag(const bf16* baseT, int ldk, int lane) {
  int n  = lane & 15;
  int kb = (lane >> 4) << 3;
  const bf16* row = baseT + n * ldk + kb;
  union { v16bf v; v8bf h[2]; } u;
  u.h[0] = *(const v8bf*)(row);
  u.h[1] = *(const v8bf*)(row + 16);
  return u.v;
}

__device__ __forceinline__ v8f wmma_bf16f32(v16bf a, v16bf b, v8f c) {
  return __builtin_amdgcn_wmma_f32_16x16x32_bf16(false, a, false, b, (short)0, c,
                                                 false, false);
}

// ---------------- small utility kernels ----------------
__global__ __launch_bounds__(256) void zero_f32_kernel(float* p, int n) {
  int i = blockIdx.x * 256 + threadIdx.x;
  if (i < n) p[i] = 0.f;
}

__global__ __launch_bounds__(256) void cast_bf16_kernel(const float* __restrict__ x,
                                                        bf16* __restrict__ y, int n) {
  int i = (blockIdx.x * 256 + threadIdx.x) * 4;
  if (i + 3 < n) {
    float4 f = *(const float4*)(x + i);
    y[i + 0] = (bf16)f.x; y[i + 1] = (bf16)f.y;
    y[i + 2] = (bf16)f.z; y[i + 3] = (bf16)f.w;
  }
}

// relayout conv weights [O][C][9] (f32) -> tap-major bf16 [9][O][C]
// so B tiles are 16B-contiguous for async_to_lds staging.
__global__ __launch_bounds__(256)
void conv_w_relayout_kernel(const float* __restrict__ w, bf16* __restrict__ out) {
  int idx = blockIdx.x * 256 + threadIdx.x;     // 0 .. 9*256*256-1
  if (idx < 9 * C_ * C_) {
    int c   = idx & 255;
    int o   = (idx >> 8) & 255;
    int tap = idx >> 16;
    out[idx] = (bf16)w[((size_t)(o * C_) + c) * 9 + tap];
  }
}

// ---------------- 3x3 conv as implicit GEMM (WMMA) + BN batch stats ----------------
// grid: x = (W/64)*(C/128)=8, y = H, z = B; block 256 (8 waves)
// WG tile: 64 pixels x 128 out-channels. Wave: 16px x 64och (4 accums).
__global__ __launch_bounds__(256)
void conv3x3_wmma_kernel(const bf16* __restrict__ xbf, const bf16* __restrict__ wcb,
                         bf16* __restrict__ yout, float* __restrict__ stats /*[2C]*/) {
  int tw = blockIdx.x & 3;
  int ot = blockIdx.x >> 2;
  int h  = blockIdx.y;
  int bb = blockIdx.z;
  int w0 = tw * 64;
  int o0 = ot * 128;
  int tid  = threadIdx.x;
  int lane = tid & 31;
  int wave = tid >> 5;
  int wpx  = wave & 3;
  int woc  = wave >> 2;          // 0/1 -> och half of 64

  __shared__ __align__(16) bf16 As[64 * 32];    // [px][k]
  __shared__ __align__(16) bf16 Bs[128 * 32];   // BT: [och][k]

  v8f acc[4];
  #pragma unroll
  for (int a = 0; a < 4; ++a) acc[a] = (v8f){0, 0, 0, 0, 0, 0, 0, 0};

  for (int tap = 0; tap < 9; ++tap) {
    int dy = tap / 3 - 1, dx = tap % 3 - 1;
    int hh = h + dy;
    bool hok = (unsigned)hh < (unsigned)H_;
    for (int c0 = 0; c0 < C_; c0 += 32) {
      // ---- B tile: async global->LDS (16B chunks, both sides contiguous) ----
#if HAVE_ASYNC_LDS
      #pragma unroll
      for (int j = 0; j < 2; ++j) {
        int cid = tid + j * 256;                 // 0..511
        int o = cid >> 2, qq = cid & 3;
        async_copy_b128(wcb + ((size_t)(tap * C_) + o0 + o) * C_ + c0 + qq * 8,
                        Bs + o * 32 + qq * 8);
      }
#else
      #pragma unroll
      for (int i = 0; i < 16; ++i) {
        int idx = tid + i * 256;
        int k = idx & 31, o = idx >> 5;
        Bs[o * 32 + k] = wcb[((size_t)(tap * C_) + o0 + o) * C_ + c0 + k];
      }
#endif
      // ---- A tile: predicated (zero-pad) synchronous staging ----
      #pragma unroll
      for (int i = 0; i < 8; ++i) {
        int idx = tid + i * 256;
        int px = idx & 63, k = idx >> 6;
        int ww = w0 + px + dx;
        bf16 val = (bf16)0.0f;
        if (hok && (unsigned)ww < (unsigned)W_)
          val = xbf[((size_t)(bb * C_ + c0 + k) * HW_) + hh * W_ + ww];
        As[px * 32 + k] = val;
      }
      async_join();
      __syncthreads();
      v16bf af = load_a_frag(As + wpx * 16 * 32, 32, lane);
      #pragma unroll
      for (int nt = 0; nt < 4; ++nt) {
        v16bf bf_ = load_b_frag(Bs + (woc * 64 + nt * 16) * 32, 32, lane);
        acc[nt] = wmma_bf16f32(af, bf_, acc[nt]);
      }
      __syncthreads();
    }
  }

  // epilogue: raw conv out (bf16) + per-channel sum/sumsq for batch-norm
  int hf = lane >> 4, n = lane & 15;
  #pragma unroll
  for (int a = 0; a < 4; ++a) {
    int ch = o0 + woc * 64 + a * 16 + n;
    float s = 0, q = 0;
    #pragma unroll
    for (int j = 0; j < 8; ++j) {
      int px = w0 + wpx * 16 + j + 8 * hf;
      float f = acc[a][j];
      yout[((size_t)(bb * C_ + ch) * HW_) + h * W_ + px] = (bf16)f;
      s += f; q += f * f;
    }
    atomicAdd(&stats[ch], s);
    atomicAdd(&stats[C_ + ch], q);
  }
}

// ---------------- fused BN(batch stats) -> relu6 -> depthwise 3x3 + channel-mean ----------------
__global__ __launch_bounds__(256)
void bn_relu6_dw_kernel(const bf16* __restrict__ y, const float* __restrict__ stats,
                        const float* __restrict__ g, const float* __restrict__ be,
                        const float* __restrict__ wdw, bf16* __restrict__ map,
                        float* __restrict__ scal /*[B*HW]*/) {
  size_t idx = (size_t)blockIdx.x * 256 + threadIdx.x;   // over B*C*H*W = 2^25
  int w = (int)(idx & 255);
  int h = (int)((idx >> 8) & 255);
  int c = (int)((idx >> 16) & 255);
  int bb = (int)(idx >> 24);

  const float Ninv = 1.0f / (float)(B_ * HW_);
  float mean = stats[c] * Ninv;
  float var  = stats[C_ + c] * Ninv - mean * mean;
  float sc   = g[c] * rsqrtf(var + 1e-5f);
  float sh   = be[c] - mean * sc;

  float acc = 0.f;
  #pragma unroll
  for (int tap = 0; tap < 9; ++tap) {
    int hh = h + tap / 3 - 1;
    int ww = w + tap % 3 - 1;
    if ((unsigned)hh < (unsigned)H_ && (unsigned)ww < (unsigned)W_) {
      float v = (float)y[((size_t)(bb * C_ + c) * HW_) + hh * W_ + ww];
      v = v * sc + sh;
      v = fminf(fmaxf(v, 0.f), 6.f);
      acc += wdw[c * 9 + tap] * v;
    }
  }
  map[idx] = (bf16)acc;
  atomicAdd(&scal[(size_t)bb * HW_ + h * W_ + w], acc * (1.0f / (float)C_));
}

// ---------------- 1x1 conv as GEMM (WMMA): out[o,p] = sum_c W[o,c]*in[c,p] ----------------
// grid: x = (HW/64)*(C/128) = 2048, y = B; block 256 (8 waves)
// Weights pre-converted to bf16 (Wb row-major [o][c]) -> async B staging.
template <bool OUT_F32>
__global__ __launch_bounds__(256)
void gemm1x1_wmma_kernel(const bf16* __restrict__ in, const bf16* __restrict__ Wb,
                         void* __restrict__ outp) {
  int pt = blockIdx.x >> 1;
  int ot = blockIdx.x & 1;
  int bb = blockIdx.y;
  int p0 = pt * 64;
  int o0 = ot * 128;
  int tid  = threadIdx.x;
  int lane = tid & 31;
  int wave = tid >> 5;
  int wpx  = wave & 3;
  int woc  = wave >> 2;

  __shared__ __align__(16) bf16 As[64 * 32];
  __shared__ __align__(16) bf16 Bs[128 * 32];

  v8f acc[4];
  #pragma unroll
  for (int a = 0; a < 4; ++a) acc[a] = (v8f){0, 0, 0, 0, 0, 0, 0, 0};

  for (int c0 = 0; c0 < C_; c0 += 32) {
    if (c0 + 32 < C_)   // speculative prefetch of next A chunk -> global_prefetch_b8
      __builtin_prefetch(in + ((size_t)(bb * C_ + c0 + 32) * HW_) + p0, 0, 1);
#if HAVE_ASYNC_LDS
    #pragma unroll
    for (int j = 0; j < 2; ++j) {
      int cid = tid + j * 256;
      int o = cid >> 2, qq = cid & 3;
      async_copy_b128(Wb + (size_t)(o0 + o) * C_ + c0 + qq * 8,
                      Bs + o * 32 + qq * 8);
    }
#else
    #pragma unroll
    for (int i = 0; i < 16; ++i) {
      int idx = tid + i * 256;
      int k = idx & 31, o = idx >> 5;
      Bs[o * 32 + k] = Wb[(size_t)(o0 + o) * C_ + c0 + k];
    }
#endif
    #pragma unroll
    for (int i = 0; i < 8; ++i) {
      int idx = tid + i * 256;
      int m = idx & 63, k = idx >> 6;
      As[m * 32 + k] = in[((size_t)(bb * C_ + c0 + k) * HW_) + p0 + m];
    }
    async_join();
    __syncthreads();
    v16bf af = load_a_frag(As + wpx * 16 * 32, 32, lane);
    #pragma unroll
    for (int nt = 0; nt < 4; ++nt) {
      v16bf bf_ = load_b_frag(Bs + (woc * 64 + nt * 16) * 32, 32, lane);
      acc[nt] = wmma_bf16f32(af, bf_, acc[nt]);
    }
    __syncthreads();
  }

  int hf = lane >> 4, n = lane & 15;
  #pragma unroll
  for (int a = 0; a < 4; ++a) {
    int ch = o0 + woc * 64 + a * 16 + n;
    #pragma unroll
    for (int j = 0; j < 8; ++j) {
      int m = j + 8 * hf;
      size_t gi = ((size_t)(bb * C_ + ch) * HW_) + p0 + m;
      if (OUT_F32) ((float*)outp)[gi] = acc[a][j];
      else         ((bf16*)outp)[gi]  = (bf16)acc[a][j];
    }
  }
}

// ---------------- window attention (8x8 windows, heads=8, d=32), WMMA ----------------
// grid: (1024 windows, 8 heads, B); block 128 (4 waves). Wave handles 16 S-rows.
__global__ __launch_bounds__(128)
void win_attn_kernel(const bf16* __restrict__ qg, const bf16* __restrict__ kg,
                     const bf16* __restrict__ vg, const float* __restrict__ illum,
                     const float* __restrict__ noise, const float* __restrict__ temp,
                     bf16* __restrict__ outg) {
  int win  = blockIdx.x;
  int head = blockIdx.y;
  int bb   = blockIdx.z;
  int wy = win >> 5, wx = win & 31;
  int tid  = threadIdx.x;
  int lane = tid & 31;
  int wave = tid >> 5;

  __shared__ __align__(16) bf16 Qs[64 * 32];     // [tok][d]
  __shared__ __align__(16) bf16 Ks[64 * 32];     // [tok][d] (BT for q@k^T)
  __shared__ __align__(16) bf16 VTs[32 * 64];    // [d][tok] (BT for P@V)
  __shared__ __align__(16) bf16 Ps[4][16 * 64];  // per-wave probs

  // ---- V^T tile: async (window row = 8 tokens = 16B, contiguous both sides) ----
#if HAVE_ASYNC_LDS
  #pragma unroll
  for (int j = 0; j < 2; ++j) {
    int cid = tid + j * 128;                     // 0..255
    int dc = cid >> 3, ty = cid & 7;
    async_copy_b128(vg + ((size_t)(bb * C_ + head * D_ + dc) * HW_) +
                        (wy * WS_ + ty) * W_ + wx * WS_,
                    VTs + dc * 64 + ty * 8);
  }
#else
  #pragma unroll
  for (int i = 0; i < 16; ++i) {
    int idx = tid + i * 128;
    int t = idx >> 5, dc = idx & 31;
    int pix = (wy * WS_ + (t >> 3)) * W_ + wx * WS_ + (t & 7);
    VTs[dc * 64 + t] = vg[((size_t)(bb * C_ + head * D_ + dc) * HW_) + pix];
  }
#endif
  // ---- q/k windows ----
  #pragma unroll
  for (int i = 0; i < 16; ++i) {
    int idx = tid + i * 128;
    int t = idx >> 5, dc = idx & 31;
    int pix = (wy * WS_ + (t >> 3)) * W_ + wx * WS_ + (t & 7);
    size_t g = ((size_t)(bb * C_ + head * D_ + dc) * HW_) + pix;
    Qs[t * 32 + dc] = qg[g];
    Ks[t * 32 + dc] = kg[g];
  }
  async_join();
  __syncthreads();

  // row-normalize q,k; fold illum/noise scalars, d^-1/2 and temperature into rows
  if (tid < 64) {
    int t = tid;
    float sq = 0.f, sk = 0.f;
    #pragma unroll
    for (int dc = 0; dc < 32; ++dc) {
      float q = (float)Qs[t * 32 + dc]; sq += q * q;
      float k = (float)Ks[t * 32 + dc]; sk += k * k;
    }
    int pix = (wy * WS_ + (t >> 3)) * W_ + wx * WS_ + (t & 7);
    float il = illum[(size_t)bb * HW_ + pix];
    float no = noise[(size_t)bb * HW_ + pix];
    float qf = (1.0f / fmaxf(sqrtf(sq), 1e-12f)) * (1.0f + il) *
               0.1767766952966369f * temp[head];
    float kf = (1.0f / fmaxf(sqrtf(sk), 1e-12f)) *
               fminf(fmaxf(1.0f - no, 0.f), 1.f);
    #pragma unroll
    for (int dc = 0; dc < 32; ++dc) {
      Qs[t * 32 + dc] = (bf16)((float)Qs[t * 32 + dc] * qf);
      Ks[t * 32 + dc] = (bf16)((float)Ks[t * 32 + dc] * kf);
    }
  }
  __syncthreads();

  // S = q @ k^T : wave computes rows [wave*16, wave*16+16) x all 64 cols
  v16bf aq = load_a_frag(Qs + wave * 16 * 32, 32, lane);
  v8f s[4];
  #pragma unroll
  for (int nt = 0; nt < 4; ++nt) {
    v8f z = {0, 0, 0, 0, 0, 0, 0, 0};
    v16bf bk = load_b_frag(Ks + nt * 16 * 32, 32, lane);
    s[nt] = wmma_bf16f32(aq, bk, z);
  }

  // softmax over 64 cols: row elems live across 16 lanes x 4 col-tiles
  int hf = lane >> 4, n = lane & 15;
  #pragma unroll
  for (int j = 0; j < 8; ++j) {
    float m = fmaxf(fmaxf(s[0][j], s[1][j]), fmaxf(s[2][j], s[3][j]));
    #pragma unroll
    for (int off = 8; off >= 1; off >>= 1) m = fmaxf(m, __shfl_xor(m, off, 32));
    float e0 = __expf(s[0][j] - m), e1 = __expf(s[1][j] - m);
    float e2 = __expf(s[2][j] - m), e3 = __expf(s[3][j] - m);
    float sum = e0 + e1 + e2 + e3;
    #pragma unroll
    for (int off = 8; off >= 1; off >>= 1) sum += __shfl_xor(sum, off, 32);
    float inv = 1.0f / sum;
    s[0][j] = e0 * inv; s[1][j] = e1 * inv;
    s[2][j] = e2 * inv; s[3][j] = e3 * inv;
  }

  // stage P (bf16) as WMMA A operand
  #pragma unroll
  for (int nt = 0; nt < 4; ++nt)
    #pragma unroll
    for (int j = 0; j < 8; ++j)
      Ps[wave][(j + 8 * hf) * 64 + nt * 16 + n] = (bf16)s[nt][j];
  __syncthreads();

  // O = P @ V : K = 64 tokens in 2 chunks; N = 32 (two 16-col tiles)
  v8f o0 = {0, 0, 0, 0, 0, 0, 0, 0};
  v8f o1 = {0, 0, 0, 0, 0, 0, 0, 0};
  #pragma unroll
  for (int kt = 0; kt < 2; ++kt) {
    v16bf ap  = load_a_frag(Ps[wave] + kt * 32, 64, lane);
    v16bf bv0 = load_b_frag(VTs + 0 * 16 * 64 + kt * 32, 64, lane);
    v16bf bv1 = load_b_frag(VTs + 1 * 16 * 64 + kt * 32, 64, lane);
    o0 = wmma_bf16f32(ap, bv0, o0);
    o1 = wmma_bf16f32(ap, bv1, o1);
  }

  #pragma unroll
  for (int j = 0; j < 8; ++j) {
    int t = wave * 16 + j + 8 * hf;
    int pix = (wy * WS_ + (t >> 3)) * W_ + wx * WS_ + (t & 7);
    outg[((size_t)(bb * C_ + head * D_ + n) * HW_) + pix]      = (bf16)o0[j];
    outg[((size_t)(bb * C_ + head * D_ + 16 + n) * HW_) + pix] = (bf16)o1[j];
  }
}

// ---------------- launcher ----------------
extern "C" void kernel_launch(void* const* d_in, const int* in_sizes, int n_in,
                              void* d_out, int out_size, void* d_ws, size_t ws_size,
                              hipStream_t stream) {
  const float* x        = (const float*)d_in[0];
  const float* w_iem    = (const float*)d_in[1];
  const float* g_iem    = (const float*)d_in[2];
  const float* b_iem    = (const float*)d_in[3];
  const float* w_nem    = (const float*)d_in[4];
  const float* g_nem    = (const float*)d_in[5];
  const float* b_nem    = (const float*)d_in[6];
  const float* w_iem_dw = (const float*)d_in[7];
  const float* w_nem_dw = (const float*)d_in[8];
  const float* w_q      = (const float*)d_in[9];
  const float* w_k      = (const float*)d_in[10];
  const float* w_v      = (const float*)d_in[11];
  const float* w_proj   = (const float*)d_in[12];
  const float* temperature = (const float*)d_in[13];

  const size_t SZ = (size_t)B_ * C_ * HW_;     // 33,554,432 elements
  bf16* xbf  = (bf16*)d_ws;                    // x in bf16
  bf16* buf0 = xbf + SZ;                       // conv_iem raw -> q
  bf16* buf1 = buf0 + SZ;                      // conv_nem raw -> k
  bf16* buf2 = buf1 + SZ;                      // illum_map    -> attn_out
  bf16* buf3 = buf2 + SZ;                      // noise_map
  bf16* buf4 = buf3 + SZ;                      // v
  float* stats_iem = (float*)(buf4 + SZ);      // [2C] sum, sumsq
  float* stats_nem = stats_iem + 2 * C_;
  float* illum_sc  = stats_nem + 2 * C_;       // [B*HW]
  float* noise_sc  = illum_sc + (size_t)B_ * HW_;
  bf16* wqb   = (bf16*)(noise_sc + (size_t)B_ * HW_);  // 16B-aligned
  bf16* wkb   = wqb + (size_t)C_ * C_;
  bf16* wvb   = wkb + (size_t)C_ * C_;
  bf16* wpb   = wvb + (size_t)C_ * C_;
  bf16* wiemb = wpb + (size_t)C_ * C_;         // tap-major [9][256][256]
  bf16* wnemb = wiemb + (size_t)9 * C_ * C_;

  // prep: zero accumulators, cast activations + weights, relayout conv weights
  int nzero = 4 * C_ + 2 * B_ * HW_;
  zero_f32_kernel<<<(nzero + 255) / 256, 256, 0, stream>>>(stats_iem, nzero);
  cast_bf16_kernel<<<(int)(SZ / 1024), 256, 0, stream>>>(x, xbf, (int)SZ);
  cast_bf16_kernel<<<64, 256, 0, stream>>>(w_q, wqb, C_ * C_);
  cast_bf16_kernel<<<64, 256, 0, stream>>>(w_k, wkb, C_ * C_);
  cast_bf16_kernel<<<64, 256, 0, stream>>>(w_v, wvb, C_ * C_);
  cast_bf16_kernel<<<64, 256, 0, stream>>>(w_proj, wpb, C_ * C_);
  conv_w_relayout_kernel<<<2304, 256, 0, stream>>>(w_iem, wiemb);
  conv_w_relayout_kernel<<<2304, 256, 0, stream>>>(w_nem, wnemb);

  dim3 cg(8, H_, B_);
  conv3x3_wmma_kernel<<<cg, 256, 0, stream>>>(xbf, wiemb, buf0, stats_iem);
  conv3x3_wmma_kernel<<<cg, 256, 0, stream>>>(xbf, wnemb, buf1, stats_nem);

  int nblk = (int)(SZ / 256);
  bn_relu6_dw_kernel<<<nblk, 256, 0, stream>>>(buf0, stats_iem, g_iem, b_iem,
                                               w_iem_dw, buf2, illum_sc);
  bn_relu6_dw_kernel<<<nblk, 256, 0, stream>>>(buf1, stats_nem, g_nem, b_nem,
                                               w_nem_dw, buf3, noise_sc);

  dim3 gg(2048, B_);
  gemm1x1_wmma_kernel<false><<<gg, 256, 0, stream>>>(buf2, wqb, (void*)buf0);
  gemm1x1_wmma_kernel<false><<<gg, 256, 0, stream>>>(buf3, wkb, (void*)buf1);
  gemm1x1_wmma_kernel<false><<<gg, 256, 0, stream>>>(xbf,  wvb, (void*)buf4);

  dim3 ag(1024, HEADS_, B_);
  win_attn_kernel<<<ag, 128, 0, stream>>>(buf0, buf1, buf4, illum_sc, noise_sc,
                                          temperature, buf2);

  gemm1x1_wmma_kernel<true><<<gg, 256, 0, stream>>>(buf2, wpb, d_out);
}
